// EncoderBlock_19275813224451
// MI455X (gfx1250) — matrix operand order repaired and checked by reading, not compiled, not run on hardware
//
#include <hip/hip_runtime.h>
#include <hip/hip_bf16.h>
#include <float.h>

// Graph-transformer encoder block for MI455X (gfx1250, wave32).
// Dense GEMMs via v_wmma_f32_16x16x32_bf16 with column-major bf16 weights in
// LDS (vectorized ds_load_b128 fragment loads). Weight/edge-tile staging uses
// CDNA5 GLOBAL_LOAD_ASYNC_TO_LDS (ASYNCcnt). Edge softmax/scatter in fp32
// with L2 atomics. Memory-bound workload (~1-1.5 GB traffic @ 23.3 TB/s).
// NOTE: tile dimension N(=100000) and E(=1200000) are multiples of 16; all
// valid tiles are full, so store paths are unguarded straight-line clauses.

typedef __bf16 bf16_t;
typedef __attribute__((ext_vector_type(16))) bf16_t v16bf;
typedef __attribute__((ext_vector_type(8)))  bf16_t v8bf;
typedef __attribute__((ext_vector_type(8)))  float  v8f;

static __device__ __forceinline__ bf16_t f2bf(float f) { return (bf16_t)f; }

static __device__ __forceinline__ v8f wmma_bf16(v16bf a, v16bf b, v8f c) {
  // (neg_a, A, neg_b, B, c_mod, C, reuse_a, reuse_b)
  return __builtin_amdgcn_wmma_f32_16x16x32_bf16(false, a, false, b, (short)0, c, false, false);
}

// ---- CDNA5 async global->LDS staging (ASYNCcnt) ----
static __device__ __forceinline__ unsigned lds_off(const void* p) {
  return (unsigned)(uintptr_t)p;  // low 32 bits of flat shared addr == LDS offset
}
static __device__ __forceinline__ void async_b128(unsigned lds, unsigned long long ga) {
  asm volatile("global_load_async_to_lds_b128 %0, %1, off" :: "v"(lds), "v"(ga) : "memory");
}
static __device__ __forceinline__ void async_b32(unsigned lds, unsigned long long ga) {
  asm volatile("global_load_async_to_lds_b32 %0, %1, off" :: "v"(lds), "v"(ga) : "memory");
}
static __device__ __forceinline__ void wait_async0() {
  asm volatile("s_wait_asynccnt 0x0" ::: "memory");
}
static __device__ __forceinline__ void wait_ds0() {
  asm volatile("s_wait_dscnt 0x0" ::: "memory");
}

// 16 contiguous bf16 -> v16bf (two ds_load_b128). p must be 16B aligned.
static __device__ __forceinline__ v16bf load_b16x16(const bf16_t* p) {
  v8bf lo = *(const v8bf*)p;
  v8bf hi = *(const v8bf*)(p + 8);
  return __builtin_shufflevector(lo, hi, 0, 1, 2, 3, 4, 5, 6, 7, 8, 9, 10, 11, 12, 13, 14, 15);
}

// A fragment: 16x32 bf16 tile (MxK), row-major LDS, leading dim ld (mult of 8).
// Lanes 0-15: M=lane, K 0..15. Lanes 16-31: M=lane-16, K 16..31.
static __device__ __forceinline__ v16bf load_a_frag_bf(const bf16_t* t, int ld, int lane) {
  int m = lane & 15, ksel = (lane >> 4) & 1;
  const bf16_t* r = t + m * ld + ksel * 16;
  v8bf lo = *(const v8bf*)r;
  v8bf hi = *(const v8bf*)(r + 8);
  return __builtin_shufflevector(lo, hi, 0, 1, 2, 3, 4, 5, 6, 7, 8, 9, 10, 11, 12, 13, 14, 15);
}

// A fragment from an f32 row-major LDS tile (convert to bf16 in regs).
static __device__ __forceinline__ v16bf load_a_frag_f32(const float* t, int ld, int lane) {
  int m = lane & 15, ksel = (lane >> 4) & 1;
  const float* r = t + m * ld + ksel * 16;
  float4 p0 = *(const float4*)(r);
  float4 p1 = *(const float4*)(r + 4);
  float4 p2 = *(const float4*)(r + 8);
  float4 p3 = *(const float4*)(r + 12);
  v16bf a;
  a[0]  = f2bf(p0.x); a[1]  = f2bf(p0.y); a[2]  = f2bf(p0.z); a[3]  = f2bf(p0.w);
  a[4]  = f2bf(p1.x); a[5]  = f2bf(p1.y); a[6]  = f2bf(p1.z); a[7]  = f2bf(p1.w);
  a[8]  = f2bf(p2.x); a[9]  = f2bf(p2.y); a[10] = f2bf(p2.z); a[11] = f2bf(p2.w);
  a[12] = f2bf(p3.x); a[13] = f2bf(p3.y); a[14] = f2bf(p3.z); a[15] = f2bf(p3.w);
  return a;
}

// Float atomic-max via signed/unsigned integer atomics (init memory to -FLT_MAX).
static __device__ __forceinline__ void atomicMaxF(float* addr, float val) {
  if (val >= 0.f) atomicMax((int*)addr, __float_as_int(val));
  else            atomicMin((unsigned int*)addr, __float_as_uint(val));
}

// ---------------------------------------------------------------------------
// Kernel 0: weights -> column-major bf16 + init segment-max / denominator.
// ---------------------------------------------------------------------------
__global__ __launch_bounds__(256) void k_prep(
    const float* __restrict__ Wq, const float* __restrict__ Wk,
    const float* __restrict__ Wv, const float* __restrict__ Wskip,
    const float* __restrict__ We, const float* __restrict__ W1,
    const float* __restrict__ W2,
    bf16_t* __restrict__ WcatT, bf16_t* __restrict__ WeT,
    bf16_t* __restrict__ W1T, bf16_t* __restrict__ W2T,
    float* __restrict__ nmax, float* __restrict__ den, int N) {
  int tid = blockIdx.x * 256 + threadIdx.x;
  if (tid < 64 * 256) {  // [64][256] -> WcatT[c*64+r]
    int r = tid >> 8, c = tid & 255;
    float val = (c < 64)  ? Wq[r * 64 + c]
              : (c < 128) ? Wk[r * 64 + c - 64]
              : (c < 192) ? Wv[r * 64 + c - 128]
                          : Wskip[r * 64 + c - 192];
    WcatT[c * 64 + r] = f2bf(val);
  }
  if (tid < 32 * 64) {   // We [32][64] -> WeT[c*32+r]
    int r = tid >> 6, c = tid & 63;
    WeT[c * 32 + r] = f2bf(We[tid]);
  }
  if (tid < 64 * 128) {  // W1 [64][128] -> W1T[c*64+r]
    int r = tid >> 7, c = tid & 127;
    W1T[c * 64 + r] = f2bf(W1[tid]);
  }
  if (tid < 128 * 64) {  // W2 [128][64] -> W2T[c*128+r]
    int r = tid >> 6, c = tid & 63;
    W2T[c * 128 + r] = f2bf(W2[tid]);
  }
  if (tid < N * 4) { nmax[tid] = -FLT_MAX; den[tid] = 0.f; }
}

// ---------------------------------------------------------------------------
// Kernel 1: LN1 + fused QKV+skip projection. One wave per 16-node tile.
//   q,k,v <- LN1(x) @ [Wq|Wk|Wv] + b ;  base <- x + LN1(x)@Wskip + bskip
// ---------------------------------------------------------------------------
__global__ __launch_bounds__(256) void k_node_pre(
    const float* __restrict__ x, const float* __restrict__ ln1w,
    const float* __restrict__ ln1b,
    const float* __restrict__ bq, const float* __restrict__ bk,
    const float* __restrict__ bv, const float* __restrict__ bskip,
    const bf16_t* __restrict__ WcatT,
    float* __restrict__ q, float* __restrict__ k, float* __restrict__ v,
    float* __restrict__ base, int N, int NT) {
  __shared__ alignas(16) bf16_t sWT[256 * 64]; // 32 KB col-major [col][k]
  __shared__ float  sB[256];
  __shared__ alignas(16) bf16_t sY[8][16 * 64];

  int tid = threadIdx.x;
  {  // async-stage weights: 2048 x 16B chunks
    unsigned lb = lds_off(sWT);
    unsigned long long gb = (unsigned long long)(uintptr_t)WcatT;
#pragma unroll
    for (int t = 0; t < 8; ++t) {
      int ch = tid + t * 256;
      async_b128(lb + ch * 16, gb + (unsigned long long)ch * 16);
    }
  }
  {
    int c = tid;
    sB[c] = (c < 64) ? bq[c] : (c < 128) ? bk[c - 64]
          : (c < 192) ? bv[c - 128] : bskip[c - 192];
  }

  int wave = tid >> 5, lane = tid & 31;
  int tile = blockIdx.x * 8 + wave;
  int node0 = tile * 16;
  bool valid = tile < NT;

  if (valid) {
    // ---- LayerNorm1: 2 lanes per row, 32 features each ----
    int r = lane & 15, hf = lane >> 4;
    int row = node0 + r; if (row >= N) row = N - 1;
    const float4* xp = (const float4*)(x + (size_t)row * 64 + hf * 32);
    float4 xv[8];
    float s1 = 0.f, s2 = 0.f;
#pragma unroll
    for (int t = 0; t < 8; ++t) {
      xv[t] = xp[t];
      s1 += xv[t].x + xv[t].y + xv[t].z + xv[t].w;
      s2 += xv[t].x * xv[t].x + xv[t].y * xv[t].y + xv[t].z * xv[t].z + xv[t].w * xv[t].w;
    }
    s1 += __shfl_xor(s1, 16, 32);
    s2 += __shfl_xor(s2, 16, 32);
    float mean = s1 * (1.f / 64.f);
    float var  = s2 * (1.f / 64.f) - mean * mean;
    float rstd = rsqrtf(var + 1e-5f);
    bf16_t* yrow = &sY[wave][r * 64 + hf * 32];
#pragma unroll
    for (int t = 0; t < 8; ++t) {
      int jb = hf * 32 + t * 4;
      yrow[t * 4 + 0] = f2bf((xv[t].x - mean) * rstd * ln1w[jb + 0] + ln1b[jb + 0]);
      yrow[t * 4 + 1] = f2bf((xv[t].y - mean) * rstd * ln1w[jb + 1] + ln1b[jb + 1]);
      yrow[t * 4 + 2] = f2bf((xv[t].z - mean) * rstd * ln1w[jb + 2] + ln1b[jb + 2]);
      yrow[t * 4 + 3] = f2bf((xv[t].w - mean) * rstd * ln1w[jb + 3] + ln1b[jb + 3]);
    }
  }
  wait_async0();
  __syncthreads();
  if (!valid) return;
  wait_ds0();

  // ---- GEMM: [16x64] x [64x256], K split in two 32-steps ----
  v16bf a0 = load_a_frag_bf(&sY[wave][0],  64, lane);
  v16bf a1 = load_a_frag_bf(&sY[wave][32], 64, lane);
  int colL = lane & 15;
  int kb = lane & 16;
  int r0 = node0 + ((lane & 16) ? 8 : 0);
#pragma unroll
  for (int nt = 0; nt < 16; ++nt) {
    float bias = sB[nt * 16 + colL];
    v8f acc;
#pragma unroll
    for (int i = 0; i < 8; ++i) acc[i] = bias;
    int colW = nt * 16 + colL;
    acc = wmma_bf16(a0, load_b16x16(&sWT[colW * 64 + kb]), acc);
    acc = wmma_bf16(a1, load_b16x16(&sWT[colW * 64 + 32 + kb]), acc);
    int c2 = colW & 63;
    float* outp = (nt < 4) ? q : (nt < 8) ? k : (nt < 12) ? v : base;
    size_t o = (size_t)r0 * 64 + c2;
    if (nt < 12) {
#pragma unroll
      for (int i = 0; i < 8; ++i) outp[o + (size_t)i * 64] = acc[i];
    } else {
#pragma unroll
      for (int i = 0; i < 8; ++i) outp[o + (size_t)i * 64] = x[o + (size_t)i * 64] + acc[i];
    }
  }
}

// ---------------------------------------------------------------------------
// Kernel 2: per-edge attention logits.  e = edge_attr@We (WMMA),
// alpha[e,h] = dot(q[dst,h,:], k[src,h,:] + e[h,:]) / 4;  atomic max per dst.
// One wave per 16-edge tile; N-tile nt == head h (DH==16).
// ---------------------------------------------------------------------------
__global__ __launch_bounds__(256) void k_edge_alpha(
    const float* __restrict__ eattr, const int* __restrict__ ei,
    const bf16_t* __restrict__ WeT,
    const float* __restrict__ q, const float* __restrict__ k,
    float* __restrict__ alpha, float* __restrict__ nmax, int E, int ET) {
  __shared__ alignas(16) bf16_t sWeT[64 * 32];    // 4 KB col-major
  __shared__ alignas(16) float  sEAf[8][16 * 32]; // f32 edge_attr tiles
  __shared__ int sIdx[8][32];                     // [0..15]=src, [16..31]=dst

  int tid = threadIdx.x;
  async_b128(lds_off(sWeT) + tid * 16,
             (unsigned long long)(uintptr_t)WeT + (unsigned long long)tid * 16);

  int wave = tid >> 5, lane = tid & 31;
  int tile = blockIdx.x * 8 + wave;
  int e0 = tile * 16;
  bool valid = tile < ET;
  if (valid) {  // async-stage edge_attr tile (f32) + indices
    int r = lane >> 1, hf2 = lane & 1;
    const float* gp = eattr + (size_t)(e0 + r) * 32 + hf2 * 16;
    unsigned lb = lds_off(&sEAf[wave][r * 32 + hf2 * 16]);
#pragma unroll
    for (int t = 0; t < 4; ++t)
      async_b128(lb + t * 16, (unsigned long long)(uintptr_t)(gp + t * 4));
    int m = lane & 15;
    const int* ip = (lane < 16) ? (ei + e0 + m) : (ei + (size_t)E + e0 + m);
    async_b32(lds_off(&sIdx[wave][lane]), (unsigned long long)(uintptr_t)ip);
  }
  wait_async0();
  __syncthreads();
  if (!valid) return;

  v16bf a = load_a_frag_f32(&sEAf[wave][0], 32, lane);
  int colL = lane & 15;
  int hi = (lane >> 4) & 1;
  int kb = lane & 16;
#pragma unroll
  for (int h = 0; h < 4; ++h) {
    v8f e8 = {};
    e8 = wmma_bf16(a, load_b16x16(&sWeT[(h * 16 + colL) * 32 + kb]), e8);
    int f = h * 16 + colL;
#pragma unroll
    for (int i = 0; i < 8; ++i) {
      int m = i + 8 * hi;
      int s  = sIdx[wave][m];
      int d2 = sIdx[wave][16 + m];
      float qv = q[(size_t)d2 * 64 + f];
      float kv = k[(size_t)s  * 64 + f];
      float val = qv * (kv + e8[i]);
      val += __shfl_xor(val, 1, 32);
      val += __shfl_xor(val, 2, 32);
      val += __shfl_xor(val, 4, 32);
      val += __shfl_xor(val, 8, 32);
      if (colL == 0) {
        float av = val * 0.25f;  // 1/sqrt(DH), DH=16
        alpha[(size_t)(e0 + m) * 4 + h] = av;
        atomicMaxF(&nmax[(size_t)d2 * 4 + h], av);
      }
    }
  }
}

// ---------------------------------------------------------------------------
// Kernel 3: ex = exp(alpha - max[dst]); den[dst] += ex. One edge per thread.
// ---------------------------------------------------------------------------
__global__ __launch_bounds__(256) void k_edge_exp(
    const int* __restrict__ ei, float* __restrict__ alpha,
    const float* __restrict__ nmax, float* __restrict__ den, int E) {
  int e = blockIdx.x * 256 + threadIdx.x;
  if (e >= E) return;
  int d2 = ei[(size_t)E + e];
  float4 a4 = *(const float4*)(alpha + (size_t)e * 4);
  float4 m4 = *(const float4*)(nmax + (size_t)d2 * 4);
  a4.x = __expf(a4.x - m4.x);
  a4.y = __expf(a4.y - m4.y);
  a4.z = __expf(a4.z - m4.z);
  a4.w = __expf(a4.w - m4.w);
  *(float4*)(alpha + (size_t)e * 4) = a4;
  float* dp = den + (size_t)d2 * 4;
  atomicAdd(dp + 0, a4.x);
  atomicAdd(dp + 1, a4.y);
  atomicAdd(dp + 2, a4.z);
  atomicAdd(dp + 3, a4.w);
}

// ---------------------------------------------------------------------------
// Kernel 4: recompute e via WMMA, msg=(v[src]+e)*attn, atomicAdd into base[dst].
// ---------------------------------------------------------------------------
__global__ __launch_bounds__(256) void k_edge_aggr(
    const float* __restrict__ eattr, const int* __restrict__ ei,
    const bf16_t* __restrict__ WeT,
    const float* __restrict__ v, const float* __restrict__ alpha,
    const float* __restrict__ den, float* __restrict__ base, int E, int ET) {
  __shared__ alignas(16) bf16_t sWeT[64 * 32];
  __shared__ alignas(16) float  sEAf[8][16 * 32];
  __shared__ int   sIdx[8][32];
  __shared__ float sAttn[8][64];  // [m*4+h]

  int tid = threadIdx.x;
  async_b128(lds_off(sWeT) + tid * 16,
             (unsigned long long)(uintptr_t)WeT + (unsigned long long)tid * 16);

  int wave = tid >> 5, lane = tid & 31;
  int tile = blockIdx.x * 8 + wave;
  int e0 = tile * 16;
  bool valid = tile < ET;
  if (valid) {
    int r = lane >> 1, hf2 = lane & 1;
    const float* gp = eattr + (size_t)(e0 + r) * 32 + hf2 * 16;
    unsigned lb = lds_off(&sEAf[wave][r * 32 + hf2 * 16]);
#pragma unroll
    for (int t = 0; t < 4; ++t)
      async_b128(lb + t * 16, (unsigned long long)(uintptr_t)(gp + t * 4));
    int m = lane & 15;
    const int* ip = (lane < 16) ? (ei + e0 + m) : (ei + (size_t)E + e0 + m);
    async_b32(lds_off(&sIdx[wave][lane]), (unsigned long long)(uintptr_t)ip);
  }
  wait_async0();
  __syncthreads();
  if (!valid) return;

  // stage attn = ex / den[dst] for all (m,h)
  for (int j = lane; j < 64; j += 32) {
    int m = j >> 2, h = j & 3;
    int d2 = sIdx[wave][16 + m];
    sAttn[wave][j] = alpha[(size_t)(e0 + m) * 4 + h] / den[(size_t)d2 * 4 + h];
  }
  wait_ds0();

  v16bf a = load_a_frag_f32(&sEAf[wave][0], 32, lane);
  int colL = lane & 15;
  int hi = (lane >> 4) & 1;
  int kb = lane & 16;
#pragma unroll
  for (int h = 0; h < 4; ++h) {
    v8f e8 = {};
    e8 = wmma_bf16(a, load_b16x16(&sWeT[(h * 16 + colL) * 32 + kb]), e8);
    int f = h * 16 + colL;
#pragma unroll
    for (int i = 0; i < 8; ++i) {
      int m = i + 8 * hi;
      int s  = sIdx[wave][m];
      int d2 = sIdx[wave][16 + m];
      float msg = (v[(size_t)s * 64 + f] + e8[i]) * sAttn[wave][m * 4 + h];
      atomicAdd(&base[(size_t)d2 * 64 + f], msg);
    }
  }
}

// ---------------------------------------------------------------------------
// Kernel 5: LN2 + MLP (GELU exact) + residual. out = base + W2@gelu(W1@LN2(base)).
// 128 threads = 4 waves, one 16-node tile per wave.
// ---------------------------------------------------------------------------
__global__ __launch_bounds__(128) void k_node_post(
    const float* __restrict__ base, const float* __restrict__ ln2w,
    const float* __restrict__ ln2b,
    const bf16_t* __restrict__ W1T, const bf16_t* __restrict__ W2T,
    const float* __restrict__ b1, const float* __restrict__ b2,
    float* __restrict__ out, int N, int NT) {
  __shared__ alignas(16) bf16_t sW1T[128 * 64];  // col-major [col][k], 16 KB
  __shared__ alignas(16) bf16_t sW2T[64 * 128];  // col-major [col][k], 16 KB
  __shared__ float sB1[128];
  __shared__ float sB2[64];
  __shared__ alignas(16) bf16_t sH[4][16 * 64];
  __shared__ alignas(16) bf16_t sT[4][16 * 128];

  int tid = threadIdx.x;
  {  // async-stage both weight matrices: 2 x 1024 x 16B chunks
    unsigned l1 = lds_off(sW1T), l2 = lds_off(sW2T);
    unsigned long long g1 = (unsigned long long)(uintptr_t)W1T;
    unsigned long long g2 = (unsigned long long)(uintptr_t)W2T;
#pragma unroll
    for (int t = 0; t < 8; ++t) {
      int ch = tid + t * 128;
      async_b128(l1 + ch * 16, g1 + (unsigned long long)ch * 16);
      async_b128(l2 + ch * 16, g2 + (unsigned long long)ch * 16);
    }
  }
  sB1[tid] = b1[tid];
  if (tid < 64) sB2[tid] = b2[tid];

  int wave = tid >> 5, lane = tid & 31;
  int tile = blockIdx.x * 4 + wave;
  int node0 = tile * 16;
  bool valid = tile < NT;

  if (valid) {  // ---- LayerNorm2 on x_new (== base after aggregation) ----
    int r = lane & 15, hf = lane >> 4;
    int row = node0 + r; if (row >= N) row = N - 1;
    const float4* xp = (const float4*)(base + (size_t)row * 64 + hf * 32);
    float4 xv[8];
    float s1 = 0.f, s2 = 0.f;
#pragma unroll
    for (int t = 0; t < 8; ++t) {
      xv[t] = xp[t];
      s1 += xv[t].x + xv[t].y + xv[t].z + xv[t].w;
      s2 += xv[t].x * xv[t].x + xv[t].y * xv[t].y + xv[t].z * xv[t].z + xv[t].w * xv[t].w;
    }
    s1 += __shfl_xor(s1, 16, 32);
    s2 += __shfl_xor(s2, 16, 32);
    float mean = s1 * (1.f / 64.f);
    float var  = s2 * (1.f / 64.f) - mean * mean;
    float rstd = rsqrtf(var + 1e-5f);
    bf16_t* hrow = &sH[wave][r * 64 + hf * 32];
#pragma unroll
    for (int t = 0; t < 8; ++t) {
      int jb = hf * 32 + t * 4;
      hrow[t * 4 + 0] = f2bf((xv[t].x - mean) * rstd * ln2w[jb + 0] + ln2b[jb + 0]);
      hrow[t * 4 + 1] = f2bf((xv[t].y - mean) * rstd * ln2w[jb + 1] + ln2b[jb + 1]);
      hrow[t * 4 + 2] = f2bf((xv[t].z - mean) * rstd * ln2w[jb + 2] + ln2b[jb + 2]);
      hrow[t * 4 + 3] = f2bf((xv[t].w - mean) * rstd * ln2w[jb + 3] + ln2b[jb + 3]);
    }
  }
  wait_async0();
  __syncthreads();
  if (!valid) return;
  wait_ds0();

  v16bf a0 = load_a_frag_bf(&sH[wave][0],  64, lane);
  v16bf a1 = load_a_frag_bf(&sH[wave][32], 64, lane);
  int colL = lane & 15;
  int hi = (lane >> 4) & 1;
  int kb = lane & 16;

  // ---- GEMM1 [16x64]x[64x128] + GELU -> sT ----
#pragma unroll
  for (int nt = 0; nt < 8; ++nt) {
    float bias = sB1[nt * 16 + colL];
    v8f acc;
#pragma unroll
    for (int i = 0; i < 8; ++i) acc[i] = bias;
    int colW = nt * 16 + colL;
    acc = wmma_bf16(a0, load_b16x16(&sW1T[colW * 64 + kb]), acc);
    acc = wmma_bf16(a1, load_b16x16(&sW1T[colW * 64 + 32 + kb]), acc);
#pragma unroll
    for (int i = 0; i < 8; ++i) {
      float t = acc[i];
      float g = 0.5f * t * (1.f + erff(t * 0.70710678118f));  // exact GELU
      int m = i + 8 * hi;
      sT[wave][m * 128 + colW] = f2bf(g);
    }
  }
  wait_ds0();

  // ---- GEMM2 [16x128]x[128x64] + residual -> out ----
  v16bf c0 = load_a_frag_bf(&sT[wave][0],  128, lane);
  v16bf c1 = load_a_frag_bf(&sT[wave][32], 128, lane);
  v16bf c2 = load_a_frag_bf(&sT[wave][64], 128, lane);
  v16bf c3 = load_a_frag_bf(&sT[wave][96], 128, lane);
  int r0 = node0 + ((lane & 16) ? 8 : 0);
#pragma unroll
  for (int nt = 0; nt < 4; ++nt) {
    float bias = sB2[nt * 16 + colL];
    v8f acc;
#pragma unroll
    for (int i = 0; i < 8; ++i) acc[i] = bias;
    int colW = nt * 16 + colL;
    acc = wmma_bf16(c0, load_b16x16(&sW2T[colW * 128 + kb]), acc);
    acc = wmma_bf16(c1, load_b16x16(&sW2T[colW * 128 + 32 + kb]), acc);
    acc = wmma_bf16(c2, load_b16x16(&sW2T[colW * 128 + 64 + kb]), acc);
    acc = wmma_bf16(c3, load_b16x16(&sW2T[colW * 128 + 96 + kb]), acc);
    size_t o = (size_t)r0 * 64 + colW;
#pragma unroll
    for (int i = 0; i < 8; ++i) out[o + (size_t)i * 64] = base[o + (size_t)i * 64] + acc[i];
  }
}

// ---------------------------------------------------------------------------
extern "C" void kernel_launch(void* const* d_in, const int* in_sizes, int n_in,
                              void* d_out, int out_size, void* d_ws, size_t ws_size,
                              hipStream_t stream) {
  const float* x     = (const float*)d_in[0];
  const int*   ei    = (const int*)  d_in[1];
  const float* eattr = (const float*)d_in[2];
  const float* ln1w  = (const float*)d_in[3];
  const float* ln1b  = (const float*)d_in[4];
  const float* ln2w  = (const float*)d_in[5];
  const float* ln2b  = (const float*)d_in[6];
  const float* Wq    = (const float*)d_in[7];
  const float* bq    = (const float*)d_in[8];
  const float* Wk    = (const float*)d_in[9];
  const float* bk    = (const float*)d_in[10];
  const float* Wv    = (const float*)d_in[11];
  const float* bv    = (const float*)d_in[12];
  const float* We    = (const float*)d_in[13];
  const float* Wskip = (const float*)d_in[14];
  const float* bskip = (const float*)d_in[15];
  const float* W1    = (const float*)d_in[16];
  const float* b1    = (const float*)d_in[17];
  const float* W2    = (const float*)d_in[18];
  const float* b2    = (const float*)d_in[19];

  const int N = in_sizes[0] / 64;
  const int E = in_sizes[2] / 32;

  char* p = (char*)d_ws;
  auto take = [&](size_t bytes) -> char* {
    char* r = p;
    p += (bytes + 255) & ~(size_t)255;
    return r;
  };
  float* q     = (float*)take((size_t)N * 64 * 4);
  float* k     = (float*)take((size_t)N * 64 * 4);
  float* v     = (float*)take((size_t)N * 64 * 4);
  float* base  = (float*)take((size_t)N * 64 * 4);
  float* alpha = (float*)take((size_t)E * 4 * 4);
  float* nmax  = (float*)take((size_t)N * 4 * 4);
  float* den   = (float*)take((size_t)N * 4 * 4);
  bf16_t* WcatT = (bf16_t*)take(64 * 256 * 2);
  bf16_t* WeT   = (bf16_t*)take(32 * 64 * 2);
  bf16_t* W1T   = (bf16_t*)take(64 * 128 * 2);
  bf16_t* W2T   = (bf16_t*)take(128 * 64 * 2);

  const int NT = (N + 15) / 16;
  const int ET = (E + 15) / 16;

  k_prep<<<(N * 4 + 255) / 256, 256, 0, stream>>>(
      Wq, Wk, Wv, Wskip, We, W1, W2, WcatT, WeT, W1T, W2T, nmax, den, N);
  k_node_pre<<<(NT + 7) / 8, 256, 0, stream>>>(
      x, ln1w, ln1b, bq, bk, bv, bskip, WcatT, q, k, v, base, N, NT);
  k_edge_alpha<<<(ET + 7) / 8, 256, 0, stream>>>(
      eattr, ei, WeT, q, k, alpha, nmax, E, ET);
  k_edge_exp<<<(E + 255) / 256, 256, 0, stream>>>(ei, alpha, nmax, den, E);
  k_edge_aggr<<<(ET + 7) / 8, 256, 0, stream>>>(
      eattr, ei, WeT, v, alpha, den, base, E, ET);
  k_node_post<<<(NT + 3) / 4, 128, 0, stream>>>(
      base, ln2w, ln2b, W1T, W2T, b1, b2, (float*)d_out, N, NT);
}
